// CrossAttentionBlock_35905926594926
// MI455X (gfx1250) — matrix operand order repaired and checked by reading, not compiled, not run on hardware
//
#include <hip/hip_runtime.h>
#include <hip/hip_bf16.h>
#include <math.h>

typedef __attribute__((ext_vector_type(16))) _Float16 v16h;
typedef __attribute__((ext_vector_type(8)))  _Float16 v8h;
typedef __attribute__((ext_vector_type(8)))  float    v8f;

#define HDIM 256
#define NH 4
#define HD 64
#define INV_TEMP 2.0f
#define LN_EPS 1e-5f

// ---------------------------------------------------------------------------
// WMMA fragment helpers (wave32; A/B share this code's contiguous k order, C/D per ISA)
// A (16x32 f16): lane l row M=(l&15), halves -> K = k0 + (l>=16?16:0) + h.
//                One contiguous 32B load.
// B (32x16 f16): lane l col N=(l&15), halves -> K = k0 + (l>=16?16:0) + h.
//                One contiguous 32B load from an [N][K] row-major matrix.
// C/D (16x16 f32): lane l col N=(l&15); VGPR r -> row M = r + (l>=16?8:0).
// ---------------------------------------------------------------------------
__device__ __forceinline__ v16h load_frag_a(const _Float16* __restrict__ base,
                                            int ld, int k0) {
  int lane = threadIdx.x & 31;
  int row  = lane & 15;
  int hi   = lane >> 4;
  const _Float16* p = base + (size_t)row * ld + k0 + hi * 16;
  return *(const v16h*)p;
}

__device__ __forceinline__ v16h load_frag_b(const _Float16* __restrict__ base,
                                            int ld, int n0, int k0) {
  int lane = threadIdx.x & 31;
  int n    = n0 + (lane & 15);
  int koff = k0 + (lane >> 4) * 16;
  return *(const v16h*)(base + (size_t)n * ld + koff);
}

__device__ __forceinline__ v8f wmma_f16(v16h a, v16h b, v8f c) {
  return __builtin_amdgcn_wmma_f32_16x16x32_f16(false, a, false, b,
                                                (short)0, c, false, false);
}

// ---------------------------------------------------------------------------
// fp32 -> fp16 conversion
// ---------------------------------------------------------------------------
__global__ void cvt_f32_f16_kernel(const float* __restrict__ src,
                                   _Float16* __restrict__ dst, int n) {
  int i = blockIdx.x * blockDim.x + threadIdx.x;
  if (i < n) dst[i] = (_Float16)src[i];
}

// ---------------------------------------------------------------------------
// QKV projection: Y = X @ W^T + b, X f16 [BS][256], W f16 [256(out)][256(k)].
// Output per head: Q/K -> [b][h][S][64]; V -> transposed [b][h][64][S].
// 128 threads = 4 waves; wave w computes cols [64w, 64w+64) for 16 rows.
// ---------------------------------------------------------------------------
__global__ void qkv_proj_kernel(const _Float16* __restrict__ X,
                                const _Float16* __restrict__ W,
                                const float* __restrict__ bias,
                                _Float16* __restrict__ out,
                                int S, int transposeV) {
  int lane = threadIdx.x & 31;
  int wave = threadIdx.x >> 5;
  int row0 = blockIdx.x * 16;
  int col0 = wave * 64;
  v8f acc[4] = {};
#pragma unroll
  for (int k0 = 0; k0 < HDIM; k0 += 32) {
    v16h a = load_frag_a(X + (size_t)row0 * HDIM, HDIM, k0);
#pragma unroll
    for (int t = 0; t < 4; ++t) {
      v16h b = load_frag_b(W, HDIM, col0 + t * 16, k0);
      acc[t] = wmma_f16(a, b, acc[t]);
    }
  }
  int rowoff = (lane >> 4) * 8;
#pragma unroll
  for (int t = 0; t < 4; ++t) {
    int col = col0 + t * 16 + (lane & 15);
    int h = col >> 6, hd = col & 63;
    float bv = bias[col];
#pragma unroll
    for (int r = 0; r < 8; ++r) {
      int row = row0 + rowoff + r;
      int bb = row / S, s = row % S;
      size_t idx = transposeV
          ? ((((size_t)bb * NH + h) * HD + hd) * S + s)
          : ((((size_t)bb * NH + h) * S + s) * HD + hd);
      out[idx] = (_Float16)(acc[t][r] + bv);
    }
  }
}

// ---------------------------------------------------------------------------
// Fused 4-head attention, 1 WG per (batch, 16-query tile), wave = head.
// Pass 1: flash (online softmax) -> ctx f16 [b][Sq][256].
// Pass 2: recompute scores, write head-averaged weights to Aout fp32.
// ---------------------------------------------------------------------------
__global__ void attention_kernel(const _Float16* __restrict__ Q,   // [B][NH][Sq][64]
                                 const _Float16* __restrict__ K,   // [B][NH][Skv][64]
                                 const _Float16* __restrict__ Vt,  // [B][NH][64][Skv]
                                 _Float16* __restrict__ ctx,       // [B][Sq][256]
                                 float* __restrict__ Aout,         // [B][Sq][Skv]
                                 int Sq, int Skv) {
  __shared__ _Float16 lds_p[NH][16 * 32];
  __shared__ float    lds_w[NH][16 * 32];
  int lane = threadIdx.x & 31;
  int head = threadIdx.x >> 5;
  int nqt  = Sq >> 4;
  int b    = blockIdx.x / nqt;
  int q0   = (blockIdx.x % nqt) << 4;

  const _Float16* Qh = Q  + (((size_t)b * NH + head) * Sq + q0) * HD;
  const _Float16* Kh = K  + (((size_t)b * NH + head) * Skv) * HD;
  const _Float16* Vh = Vt + (((size_t)b * NH + head) * HD) * Skv;

  v16h aq0 = load_frag_a(Qh, HD, 0);
  v16h aq1 = load_frag_a(Qh, HD, 32);

  v8f  o[4] = {};
  float m[8], ell[8];
#pragma unroll
  for (int r = 0; r < 8; ++r) { m[r] = -1e30f; ell[r] = 0.0f; }

  int rowoff = (lane >> 4) * 8;
  int coll   = lane & 15;

  for (int j0 = 0; j0 < Skv; j0 += 32) {
    __builtin_prefetch(Kh + (size_t)(j0 + 32) * HD, 0, 0);
    __builtin_prefetch(Vh + (size_t)j0 + 32, 0, 0);
    // S = (Q K^T) * (1/temp), 16q x 32kv, K-dim 64 chained over two WMMAs
    v8f s0 = {}, s1 = {};
    s0 = wmma_f16(aq0, load_frag_b(Kh + (size_t)j0 * HD, HD, 0, 0),  s0);
    s0 = wmma_f16(aq1, load_frag_b(Kh + (size_t)j0 * HD, HD, 0, 32), s0);
    s1 = wmma_f16(aq0, load_frag_b(Kh + (size_t)(j0 + 16) * HD, HD, 0, 0),  s1);
    s1 = wmma_f16(aq1, load_frag_b(Kh + (size_t)(j0 + 16) * HD, HD, 0, 32), s1);
    // online softmax: rows 0-7 live in lanes 0-15, rows 8-15 in lanes 16-31
#pragma unroll
    for (int r = 0; r < 8; ++r) {
      float v0 = s0[r] * INV_TEMP;
      float v1 = s1[r] * INV_TEMP;
      float cm = fmaxf(v0, v1);
#pragma unroll
      for (int x = 1; x < 16; x <<= 1) cm = fmaxf(cm, __shfl_xor(cm, x, 32));
      float mn   = fmaxf(m[r], cm);
      float corr = __expf(m[r] - mn);
      float p0   = __expf(v0 - mn);
      float p1   = __expf(v1 - mn);
      float ps   = p0 + p1;
#pragma unroll
      for (int x = 1; x < 16; x <<= 1) ps += __shfl_xor(ps, x, 32);
      ell[r] = ell[r] * corr + ps;
      m[r]   = mn;
#pragma unroll
      for (int t = 0; t < 4; ++t) o[t][r] *= corr;
      s0[r] = p0; s1[r] = p1;
    }
    // stage P (C-layout) -> LDS -> reload as A-layout f16
    __syncthreads();
#pragma unroll
    for (int r = 0; r < 8; ++r) {
      lds_p[head][(rowoff + r) * 32 + coll]      = (_Float16)s0[r];
      lds_p[head][(rowoff + r) * 32 + coll + 16] = (_Float16)s1[r];
    }
    __syncthreads();
    v16h ap = load_frag_a(lds_p[head], 32, 0);
#pragma unroll
    for (int t = 0; t < 4; ++t)
      o[t] = wmma_f16(ap, load_frag_b(Vh, Skv, t * 16, j0), o[t]);
  }
  // normalize and write context (merged heads, f16)
#pragma unroll
  for (int r = 0; r < 8; ++r) {
    float inv = 1.0f / ell[r];
    int q = q0 + rowoff + r;
#pragma unroll
    for (int t = 0; t < 4; ++t)
      ctx[((size_t)b * Sq + q) * HDIM + head * HD + t * 16 + coll] =
          (_Float16)(o[t][r] * inv);
  }
  // pass 2: recompute scores, write 0.25 * sum_heads softmax -> Aout
  for (int j0 = 0; j0 < Skv; j0 += 32) {
    v8f s0 = {}, s1 = {};
    s0 = wmma_f16(aq0, load_frag_b(Kh + (size_t)j0 * HD, HD, 0, 0),  s0);
    s0 = wmma_f16(aq1, load_frag_b(Kh + (size_t)j0 * HD, HD, 0, 32), s0);
    s1 = wmma_f16(aq0, load_frag_b(Kh + (size_t)(j0 + 16) * HD, HD, 0, 0),  s1);
    s1 = wmma_f16(aq1, load_frag_b(Kh + (size_t)(j0 + 16) * HD, HD, 0, 32), s1);
    __syncthreads();
#pragma unroll
    for (int r = 0; r < 8; ++r) {
      float inv = 0.25f / ell[r];
      lds_w[head][(rowoff + r) * 32 + coll]      = __expf(s0[r] * INV_TEMP - m[r]) * inv;
      lds_w[head][(rowoff + r) * 32 + coll + 16] = __expf(s1[r] * INV_TEMP - m[r]) * inv;
    }
    __syncthreads();
    for (int e = threadIdx.x; e < 512; e += 128) {
      int row = e >> 5, c = e & 31;
      float sum = lds_w[0][row * 32 + c] + lds_w[1][row * 32 + c] +
                  lds_w[2][row * 32 + c] + lds_w[3][row * 32 + c];
      Aout[((size_t)b * Sq + q0 + row) * Skv + j0 + c] = sum;
    }
  }
}

// ---------------------------------------------------------------------------
// Output projection + bias + ReLU + LayerNorm, fused. 16 rows x 256 cols / WG.
// ---------------------------------------------------------------------------
__global__ void out_proj_ln_kernel(const _Float16* __restrict__ ctx, // [BS][256]
                                   const _Float16* __restrict__ Wo,  // [256][256]
                                   const float* __restrict__ bias,
                                   const float* __restrict__ gamma,
                                   const float* __restrict__ beta,
                                   float* __restrict__ out) {
  __shared__ float tile[16 * HDIM];
  __shared__ float stats[16 * 2];
  int lane = threadIdx.x & 31;
  int wave = threadIdx.x >> 5;
  int row0 = blockIdx.x * 16;
  int col0 = wave * 64;
  v8f acc[4] = {};
#pragma unroll
  for (int k0 = 0; k0 < HDIM; k0 += 32) {
    v16h a = load_frag_a(ctx + (size_t)row0 * HDIM, HDIM, k0);
#pragma unroll
    for (int t = 0; t < 4; ++t)
      acc[t] = wmma_f16(a, load_frag_b(Wo, HDIM, col0 + t * 16, k0), acc[t]);
  }
  int rowoff = (lane >> 4) * 8;
#pragma unroll
  for (int t = 0; t < 4; ++t) {
    int col = col0 + t * 16 + (lane & 15);
    float bv = bias[col];
#pragma unroll
    for (int r = 0; r < 8; ++r)
      tile[(rowoff + r) * HDIM + col] = fmaxf(acc[t][r] + bv, 0.0f);
  }
  __syncthreads();
  if (threadIdx.x < 16) {
    float s = 0.0f, s2 = 0.0f;
    for (int c = 0; c < HDIM; ++c) {
      float v = tile[threadIdx.x * HDIM + c];
      s += v; s2 += v * v;
    }
    float mu  = s * (1.0f / HDIM);
    float var = s2 * (1.0f / HDIM) - mu * mu;
    stats[threadIdx.x * 2]     = mu;
    stats[threadIdx.x * 2 + 1] = rsqrtf(var + LN_EPS);
  }
  __syncthreads();
  for (int e = threadIdx.x; e < 16 * HDIM; e += 128) {
    int r = e >> 8, c = e & 255;
    out[((size_t)row0 + r) * HDIM + c] =
        (tile[e] - stats[r * 2]) * stats[r * 2 + 1] * gamma[c] + beta[c];
  }
}

// ---------------------------------------------------------------------------
extern "C" void kernel_launch(void* const* d_in, const int* in_sizes, int n_in,
                              void* d_out, int out_size, void* d_ws, size_t ws_size,
                              hipStream_t stream) {
  const int B = 8, SI = 2048, ST = 1024;
  const size_t NIMG = (size_t)B * SI * HDIM;   // 4,194,304
  const size_t NTXT = (size_t)B * ST * HDIM;   // 2,097,152
  const size_t NW   = (size_t)HDIM * HDIM;     // 65,536

  const float* image = (const float*)d_in[0];
  const float* text  = (const float*)d_in[1];
  // i2t: 2 wq, 3 bq, 4 wk, 5 bk, 6 wv, 7 bv, 8 wo, 9 bo, 10 gamma, 11 beta
  // t2i: 12..21 same order
  const float* wsrc[8] = {
    (const float*)d_in[2],  (const float*)d_in[4],  (const float*)d_in[6],  (const float*)d_in[8],
    (const float*)d_in[12], (const float*)d_in[14], (const float*)d_in[16], (const float*)d_in[18]
  };

  char* wp = (char*)d_ws;
  auto alloc = [&](size_t bytes) {
    void* r = (void*)wp;
    wp += (bytes + 255) & ~(size_t)255;
    return r;
  };
  _Float16* h_img = (_Float16*)alloc(NIMG * 2);
  _Float16* h_txt = (_Float16*)alloc(NTXT * 2);
  _Float16* w16[8];
  for (int i = 0; i < 8; ++i) w16[i] = (_Float16*)alloc(NW * 2);
  _Float16* Qi  = (_Float16*)alloc(NIMG * 2);  // [8][4][2048][64]
  _Float16* Ki  = (_Float16*)alloc(NTXT * 2);  // [8][4][1024][64]
  _Float16* Vti = (_Float16*)alloc(NTXT * 2);  // [8][4][64][1024]
  _Float16* Qt  = (_Float16*)alloc(NTXT * 2);
  _Float16* Kt  = (_Float16*)alloc(NIMG * 2);
  _Float16* Vtt = (_Float16*)alloc(NIMG * 2);
  _Float16* ctx_i = (_Float16*)alloc(NIMG * 2);
  _Float16* ctx_t = (_Float16*)alloc(NTXT * 2);

  float* out_i2t  = (float*)d_out;                         // [8][2048][256]
  float* out_t2i  = out_i2t + NIMG;                        // [8][1024][256]
  float* A_i2t    = out_t2i + NTXT;                        // [8][2048][1024]
  float* A_t2i    = A_i2t + (size_t)B * SI * ST;           // [8][1024][2048]

  // 1) fp32 -> fp16 conversions
  cvt_f32_f16_kernel<<<(int)((NIMG + 255) / 256), 256, 0, stream>>>(image, h_img, (int)NIMG);
  cvt_f32_f16_kernel<<<(int)((NTXT + 255) / 256), 256, 0, stream>>>(text, h_txt, (int)NTXT);
  for (int i = 0; i < 8; ++i)
    cvt_f32_f16_kernel<<<(int)((NW + 255) / 256), 256, 0, stream>>>(wsrc[i], w16[i], (int)NW);

  // 2) QKV projections (w16: 0..3 = i2t wq,wk,wv,wo ; 4..7 = t2i)
  qkv_proj_kernel<<<B * SI / 16, 128, 0, stream>>>(h_img, w16[0], (const float*)d_in[3],  Qi,  SI, 0);
  qkv_proj_kernel<<<B * ST / 16, 128, 0, stream>>>(h_txt, w16[1], (const float*)d_in[5],  Ki,  ST, 0);
  qkv_proj_kernel<<<B * ST / 16, 128, 0, stream>>>(h_txt, w16[2], (const float*)d_in[7],  Vti, ST, 1);
  qkv_proj_kernel<<<B * ST / 16, 128, 0, stream>>>(h_txt, w16[4], (const float*)d_in[13], Qt,  ST, 0);
  qkv_proj_kernel<<<B * SI / 16, 128, 0, stream>>>(h_img, w16[5], (const float*)d_in[15], Kt,  SI, 0);
  qkv_proj_kernel<<<B * SI / 16, 128, 0, stream>>>(h_img, w16[6], (const float*)d_in[17], Vtt, SI, 1);

  // 3) fused attention (+ averaged weights straight to d_out)
  attention_kernel<<<B * SI / 16, 128, 0, stream>>>(Qi, Ki, Vti, ctx_i, A_i2t, SI, ST);
  attention_kernel<<<B * ST / 16, 128, 0, stream>>>(Qt, Kt, Vtt, ctx_t, A_t2i, ST, SI);

  // 4) output projection + ReLU + LayerNorm
  out_proj_ln_kernel<<<B * SI / 16, 128, 0, stream>>>(ctx_i, w16[3], (const float*)d_in[9],
                                                      (const float*)d_in[10], (const float*)d_in[11], out_i2t);
  out_proj_ln_kernel<<<B * ST / 16, 128, 0, stream>>>(ctx_t, w16[7], (const float*)d_in[19],
                                                      (const float*)d_in[20], (const float*)d_in[21], out_t2i);
}
